// RTM_76622216561047
// MI455X (gfx1250) — matrix operand order repaired and checked by reading, not compiled, not run on hardware
//
#include <hip/hip_runtime.h>
#include <hip/hip_bf16.h>
#include <math.h>

typedef __attribute__((ext_vector_type(16))) __bf16 v16bf;
typedef __attribute__((ext_vector_type(8)))  __bf16 v8bf;
typedef __attribute__((ext_vector_type(8)))  float  v8f;

#define A_DIM 8
#define B_DIM 1024
#define C_DIM 768
#define H_DIM 12
#define DH    64
#define NBLK  6
#define R_TOK (A_DIM * B_DIM)   /* 8192 tokens */
#define FC    3072              /* 4*C */
#define QKV3  2304              /* 3*C */
#define EPSF  1e-5f

#define KC    128               /* k-chunk staged in LDS            */
#define LDSW  136               /* padded LDS row stride (halfwords) */

// ---------- helpers ----------
static __device__ __forceinline__ unsigned short f2bf(float f) {
  unsigned int u = __float_as_uint(f);
  u += 0x7FFFu + ((u >> 16) & 1u);     // round-to-nearest-even
  return (unsigned short)(u >> 16);
}

// ---------- elementwise f32 -> bf16 ----------
__global__ void cvt_bf16_kernel(const float* __restrict__ in,
                                unsigned short* __restrict__ out, int n) {
  int i = blockIdx.x * 256 + threadIdx.x;
  if (i < n) out[i] = f2bf(in[i]);
}

// ---------- LayerNorm ----------
// MODE 0: h = LN(s)*g+b               -> bf16 only   (ln1)
// MODE 1: s = LN(t)*g+b + t           -> f32 + bf16  (ln2 + residual)
template <int MODE>
__global__ __launch_bounds__(256) void ln_kernel(const float* __restrict__ in,
                                                 const float* __restrict__ g,
                                                 const float* __restrict__ bta,
                                                 float* __restrict__ outf,
                                                 unsigned short* __restrict__ outb) {
  __shared__ float red[256];
  const int row = blockIdx.x;
  const int tid = threadIdx.x;
  const float* x = in + (size_t)row * C_DIM;
  float v0 = x[tid], v1 = x[tid + 256], v2 = x[tid + 512];

  red[tid] = v0 + v1 + v2;
  __syncthreads();
  for (int off = 128; off > 0; off >>= 1) {
    if (tid < off) red[tid] += red[tid + off];
    __syncthreads();
  }
  float mean = red[0] * (1.0f / C_DIM);
  __syncthreads();

  float d0 = v0 - mean, d1 = v1 - mean, d2 = v2 - mean;
  red[tid] = d0 * d0 + d1 * d1 + d2 * d2;
  __syncthreads();
  for (int off = 128; off > 0; off >>= 1) {
    if (tid < off) red[tid] += red[tid + off];
    __syncthreads();
  }
  float rstd = rsqrtf(red[0] * (1.0f / C_DIM) + EPSF);

  float vv[3] = {v0, v1, v2};
  #pragma unroll
  for (int p = 0; p < 3; ++p) {
    int c = tid + p * 256;
    float y = (vv[p] - mean) * rstd * g[c] + bta[c];
    size_t idx = (size_t)row * C_DIM + c;
    if (MODE == 0) {
      outb[idx] = f2bf(y);
    } else {
      float y2 = y + vv[p];
      outf[idx] = y2;
      outb[idx] = f2bf(y2);
    }
  }
}

// ---------- rsa[ab,h] = dot(q,k)/sqrt(Dh) ----------
__global__ void rsa_kernel(const float* __restrict__ qkv, float* __restrict__ rsa) {
  int i = blockIdx.x * 256 + threadIdx.x;
  if (i >= R_TOK * H_DIM) return;
  int ab = i / H_DIM, h = i % H_DIM;
  const float* q = qkv + (size_t)ab * QKV3 + h * DH;
  const float* k = q + C_DIM;
  float s = 0.f;
  #pragma unroll
  for (int d = 0; d < DH; ++d) s += q[d] * k[d];
  rsa[i] = s * 0.125f;   // 1/sqrt(64)
}

// ---------- causal cumsum over seq of rsa*v -> bf16 ----------
__global__ void cumsum_kernel(const float* __restrict__ qkv,
                              const float* __restrict__ rsa,
                              unsigned short* __restrict__ imv) {
  int a = blockIdx.x / H_DIM, h = blockIdx.x % H_DIM;
  int d = threadIdx.x;   // 0..63
  float acc = 0.f;
  for (int j = 0; j < B_DIM; ++j) {
    int ab = a * B_DIM + j;
    float r = rsa[ab * H_DIM + h];
    acc += r * qkv[(size_t)ab * QKV3 + 2 * C_DIM + h * DH + d];
    imv[(size_t)ab * C_DIM + h * DH + d] = f2bf(acc);
  }
}

// ==================== WMMA bf16 GEMM ====================
// out[r, c] = sum_k act[r,k] * W[c,k]
// Block = 256 threads (8 waves) covers 256 rows x 64 cols.
// Each wave owns a 32x64 tile: 2x4 = 8 WMMA accumulators (64 VGPRs; no
// spills). B-fragments staged in LDS via async global->LDS copies, shared
// by all 8 waves, reused 2x per wave. Double-buffered k-chunks of 128.
enum { EP_POSBIAS = 0, EP_F32 = 1, EP_RESID = 2, EP_BIAS_GELU_BF16 = 3, EP_BIAS_F32 = 4 };

template <int MODE>
__global__ __launch_bounds__(256) void gemm_wmma(const unsigned short* __restrict__ Am,
                                                 const unsigned short* __restrict__ Wm,
                                                 float* __restrict__ outf,
                                                 unsigned short* __restrict__ outb,
                                                 const float* __restrict__ bias,
                                                 const float* __restrict__ resid,
                                                 int Kd, int Cout, int tiles_c) {
  __shared__ __align__(16) unsigned short shB[2][64 * LDSW];

  const int bc   = blockIdx.x % tiles_c;
  const int br   = blockIdx.x / tiles_c;
  const int wv   = threadIdx.x >> 5;
  const int lane = threadIdx.x & 31;
  const int half = lane >> 4;          // 0/1
  const int nrow = lane & 15;          // A: M-row ; B/D: N-col
  const int col0 = bc * 64;
  const int row0 = br * 256 + wv * 32;
  const int tid  = threadIdx.x;
  const int nst  = Kd / KC;

  // async copy of one 64-row x 128-k B chunk into LDS buffer `buf`
  auto stage_B = [&](int buf, int kbase) {
    #pragma unroll
    for (int i = 0; i < 4; ++i) {
      int chunk = tid + i * 256;              // 1024 x b128 chunks
      int brow  = chunk >> 4;                 // 0..63
      int koff  = (chunk & 15) << 3;          // 0..120 (halfwords)
      const unsigned short* src = Wm + (size_t)(col0 + brow) * Kd + kbase + koff;
      unsigned lds = (unsigned)(unsigned long long)&shB[buf][brow * LDSW + koff];
      asm volatile("global_load_async_to_lds_b128 %0, %1, off"
                   :: "v"(lds), "v"(src) : "memory");
    }
  };

  v8f acc[2][4] = {};
  const unsigned short* ap = Am + (size_t)(row0 + nrow) * Kd + (half << 3);

  stage_B(0, 0);
  for (int s = 0; s < nst; ++s) {
    asm volatile("s_wait_asynccnt 0x0" ::: "memory");
    __syncthreads();
    if (s + 1 < nst) stage_B((s + 1) & 1, (s + 1) * KC);

    const int buf = s & 1;
    const int kg  = s * KC;
    #pragma unroll
    for (int kk = 0; kk < KC; kk += 32) {
      // A 16x32 bf16 fragments: lane(half,m) K = {half*8+0..7} U {16+half*8+0..7}
      v16bf a[2];
      #pragma unroll
      for (int mi = 0; mi < 2; ++mi) {
        const unsigned short* p = ap + (size_t)mi * 16 * Kd + kg + kk;
        v8bf a0 = *(const v8bf*)p;
        v8bf a1 = *(const v8bf*)(p + 16);
        #pragma unroll
        for (int i = 0; i < 8; ++i) { a[mi][i] = a0[i]; a[mi][i + 8] = a1[i]; }
      }
      #pragma unroll
      for (int ni = 0; ni < 4; ++ni) {
        // B 32x16 bf16 fragment from LDS: lane(half,n) K = {half*8+0..7} U {16+half*8+0..7}
        int boff = (ni * 16 + nrow) * LDSW + kk + (half << 3);
        v8bf b0 = *(const v8bf*)&shB[buf][boff];
        v8bf b1 = *(const v8bf*)&shB[buf][boff + 16];
        v16bf bv;
        #pragma unroll
        for (int i = 0; i < 8; ++i) { bv[i] = b0[i]; bv[i + 8] = b1[i]; }
        #pragma unroll
        for (int mi = 0; mi < 2; ++mi)
          acc[mi][ni] = __builtin_amdgcn_wmma_f32_16x16x32_bf16(
              false, a[mi], false, bv, (short)0, acc[mi][ni], false, false);
      }
    }
    __syncthreads();
  }

  // D layout: VGPR r -> M = 8*half + r, N = lane&15
  #pragma unroll
  for (int mi = 0; mi < 2; ++mi) {
    #pragma unroll
    for (int ni = 0; ni < 4; ++ni) {
      const int col = col0 + ni * 16 + nrow;
      #pragma unroll
      for (int r = 0; r < 8; ++r) {
        const int row = row0 + mi * 16 + half * 8 + r;
        const size_t idx = (size_t)row * Cout + col;
        float v = acc[mi][ni][r];
        if (MODE == EP_POSBIAS) {
          int j = row & (B_DIM - 1);
          float freq = __expf(-9.2103403720f * (float)(col >> 1) * (1.0f / 192.0f));
          float ang = (float)j * freq;
          v += (col & 1) ? cosf(ang) : sinf(ang);
          outf[idx] = v;
        } else if (MODE == EP_F32) {
          outf[idx] = v;
        } else if (MODE == EP_RESID) {
          outf[idx] = v + resid[idx];
        } else if (MODE == EP_BIAS_GELU_BF16) {
          v += bias[col];
          float gl = 0.5f * v * (1.0f + erff(v * 0.70710678f));
          outb[idx] = f2bf(gl);
        } else { // EP_BIAS_F32
          outf[idx] = v + bias[col];
        }
      }
    }
  }
}

// ---------- host ----------
static void launch_gemm(int mode, const unsigned short* Am, const unsigned short* Wm,
                        float* outf, unsigned short* outb, const float* bias,
                        const float* resid, int Kd, int Cout, hipStream_t stream) {
  int tiles_c = Cout / 64;
  int blocks = (R_TOK / 256) * tiles_c;
  switch (mode) {
    case EP_POSBIAS:
      gemm_wmma<EP_POSBIAS><<<blocks, 256, 0, stream>>>(Am, Wm, outf, outb, bias, resid, Kd, Cout, tiles_c); break;
    case EP_F32:
      gemm_wmma<EP_F32><<<blocks, 256, 0, stream>>>(Am, Wm, outf, outb, bias, resid, Kd, Cout, tiles_c); break;
    case EP_RESID:
      gemm_wmma<EP_RESID><<<blocks, 256, 0, stream>>>(Am, Wm, outf, outb, bias, resid, Kd, Cout, tiles_c); break;
    case EP_BIAS_GELU_BF16:
      gemm_wmma<EP_BIAS_GELU_BF16><<<blocks, 256, 0, stream>>>(Am, Wm, outf, outb, bias, resid, Kd, Cout, tiles_c); break;
    default:
      gemm_wmma<EP_BIAS_F32><<<blocks, 256, 0, stream>>>(Am, Wm, outf, outb, bias, resid, Kd, Cout, tiles_c); break;
  }
}

extern "C" void kernel_launch(void* const* d_in, const int* in_sizes, int n_in,
                              void* d_out, int out_size, void* d_ws, size_t ws_size,
                              hipStream_t stream) {
  const float* x      = (const float*)d_in[0];
  const float* weight = (const float*)d_in[1];
  const float* Wqkv   = (const float*)d_in[2];
  const float* Wo     = (const float*)d_in[3];
  const float* ln1_g  = (const float*)d_in[4];
  const float* ln1_b  = (const float*)d_in[5];
  const float* ln2_g  = (const float*)d_in[6];
  const float* ln2_b  = (const float*)d_in[7];
  const float* fc1_w  = (const float*)d_in[8];
  const float* fc1_b  = (const float*)d_in[9];
  const float* fc2_w  = (const float*)d_in[10];
  const float* fc2_b  = (const float*)d_in[11];

  // workspace carve-out (256B aligned)
  size_t off = 0;
  auto alloc = [&](size_t bytes) { size_t o = off; off += (bytes + 255) & ~(size_t)255; return o; };
  char* ws = (char*)d_ws;
  float*          s_f32    = (float*)(ws + alloc((size_t)R_TOK * C_DIM * 4));
  float*          t_f32    = (float*)(ws + alloc((size_t)R_TOK * C_DIM * 4));
  unsigned short* h_bf16   = (unsigned short*)(ws + alloc((size_t)R_TOK * C_DIM * 2));
  unsigned short* imv_bf16 = (unsigned short*)(ws + alloc((size_t)R_TOK * C_DIM * 2));
  float*          qkv_f32  = (float*)(ws + alloc((size_t)R_TOK * QKV3 * 4));
  float*          rsa_f32  = (float*)(ws + alloc((size_t)R_TOK * H_DIM * 4));
  unsigned short* hmlp_bf16= (unsigned short*)(ws + alloc((size_t)R_TOK * FC * 2));
  unsigned short* wb_w     = (unsigned short*)(ws + alloc((size_t)C_DIM * C_DIM * 2));
  unsigned short* wb_qkv   = (unsigned short*)(ws + alloc((size_t)NBLK * QKV3 * C_DIM * 2));
  unsigned short* wb_wo    = (unsigned short*)(ws + alloc((size_t)NBLK * C_DIM * C_DIM * 2));
  unsigned short* wb_fc1   = (unsigned short*)(ws + alloc((size_t)FC * C_DIM * 2));
  unsigned short* wb_fc2   = (unsigned short*)(ws + alloc((size_t)C_DIM * FC * 2));

  auto cvt = [&](const float* src, unsigned short* dst, int n) {
    cvt_bf16_kernel<<<(n + 255) / 256, 256, 0, stream>>>(src, dst, n);
  };

  // one-time (per launch) bf16 conversions
  cvt(x, h_bf16, R_TOK * C_DIM);
  cvt(weight, wb_w, C_DIM * C_DIM);
  cvt(Wqkv, wb_qkv, NBLK * QKV3 * C_DIM);
  cvt(Wo, wb_wo, NBLK * C_DIM * C_DIM);
  cvt(fc1_w, wb_fc1, FC * C_DIM);
  cvt(fc2_w, wb_fc2, C_DIM * FC);

  // s = x @ weight^T + pos_bias
  launch_gemm(EP_POSBIAS, h_bf16, wb_w, s_f32, nullptr, nullptr, nullptr, C_DIM, C_DIM, stream);

  for (int blk = 0; blk < NBLK; ++blk) {
    // h = LN1(s) -> bf16
    ln_kernel<0><<<R_TOK, 256, 0, stream>>>(s_f32, ln1_g, ln1_b, nullptr, h_bf16);
    // qkv = h @ Wqkv[blk]^T  (Cout = 2304)
    launch_gemm(EP_F32, h_bf16, wb_qkv + (size_t)blk * QKV3 * C_DIM,
                qkv_f32, nullptr, nullptr, nullptr, C_DIM, QKV3, stream);
    // rsa = dot(q,k)/8
    rsa_kernel<<<(R_TOK * H_DIM + 255) / 256, 256, 0, stream>>>(qkv_f32, rsa_f32);
    // imv = cumsum_j(rsa * v) -> bf16
    cumsum_kernel<<<A_DIM * H_DIM, DH, 0, stream>>>(qkv_f32, rsa_f32, imv_bf16);
    // t = imv @ Wo[blk]^T + s
    launch_gemm(EP_RESID, imv_bf16, wb_wo + (size_t)blk * C_DIM * C_DIM,
                t_f32, nullptr, nullptr, s_f32, C_DIM, C_DIM, stream);
    // s = LN2(t) + t  (f32 + bf16)
    ln_kernel<1><<<R_TOK, 256, 0, stream>>>(t_f32, ln2_g, ln2_b, s_f32, h_bf16);
    // hmlp = gelu(s @ fc1^T + b1) -> bf16
    launch_gemm(EP_BIAS_GELU_BF16, h_bf16, wb_fc1, nullptr, hmlp_bf16,
                fc1_b, nullptr, C_DIM, FC, stream);
    // s = hmlp @ fc2^T + b2  (last block writes d_out directly)
    float* dst = (blk == NBLK - 1) ? (float*)d_out : s_f32;
    launch_gemm(EP_BIAS_F32, hmlp_bf16, wb_fc2, dst, nullptr,
                fc2_b, nullptr, FC, C_DIM, stream);
  }
}
